// Agent_16432544875280
// MI455X (gfx1250) — matrix operand order repaired and checked by reading, not compiled, not run on hardware
//
#include <hip/hip_runtime.h>

typedef __attribute__((ext_vector_type(16))) _Float16 v16h;
typedef __attribute__((ext_vector_type(8)))  _Float16 v8h;
typedef __attribute__((ext_vector_type(8)))  float    v8f;
typedef __attribute__((ext_vector_type(4)))  float    v4f;
typedef __attribute__((ext_vector_type(4)))  unsigned int u32x4;
typedef __attribute__((ext_vector_type(8)))  int      i32x8;
typedef __attribute__((ext_vector_type(4)))  int      i32x4;

#define TILE_ROWS 64
// s_in: TDM-padded layout. 96 halves/row stored as 3 segments of 32 halves,
// each segment 40 halves apart (TDM pads 4 dwords after every 16 dwords).
// Row stride 120 halves = 240B (16B aligned); 60 dwords -> conflict-free banks.
#define IN_STRIDE  120
#define IN_CSTRIDE 40
#define H_STRIDE   136  // halves; 128 cols + pad; row stride 272B (16B aligned)
#define H_CSTRIDE  32
#define MU_STRIDE  80   // halves
#define W3_STRIDE  132  // floats; transposed head weights in LDS

// Packed-weight fragment table (fragment = 32 lanes x 16 halves = 512 halves)
#define NFRAG 260
#define FS_E0 0
#define FS_E1 16
#define FS_E2 48
#define FS_M0 80
#define FS_M1 88
#define FS_M2 120
#define FS_MF 152
#define FS_P0 172
#define FS_P1 196
#define FS_P2 228
#define WP_BYTES ((size_t)NFRAG * 512 * 2)

// ---------------------------------------------------------------------------
// Prep 1: repack WMMA-layer weights f32 [K][Nv] -> f16 fragments in exact
// v_wmma_f32_16x16x32_f16 B-operand lane layout.
// ---------------------------------------------------------------------------
__global__ void __launch_bounds__(256)
pack_weights_kernel(const float* __restrict__ ew0, const float* __restrict__ ew1,
                    const float* __restrict__ ew2,
                    const float* __restrict__ mw0, const float* __restrict__ mw1,
                    const float* __restrict__ mw2, const float* __restrict__ mw3,
                    const float* __restrict__ pw0, const float* __restrict__ pw1,
                    const float* __restrict__ pw2,
                    _Float16* __restrict__ wp)
{
  const int gid  = blockIdx.x * 256 + threadIdx.x;
  const int frag = gid >> 5;
  const int lane = gid & 31;
  if (frag >= NFRAG) return;

  const float* W; int K, Nv, start;
  if      (frag < FS_E1) { W = ew0; K = 64;  Nv = 128; start = FS_E0; }
  else if (frag < FS_E2) { W = ew1; K = 128; Nv = 128; start = FS_E1; }
  else if (frag < FS_M0) { W = ew2; K = 128; Nv = 128; start = FS_E2; }
  else if (frag < FS_M1) { W = mw0; K = 32;  Nv = 128; start = FS_M0; }
  else if (frag < FS_M2) { W = mw1; K = 128; Nv = 128; start = FS_M1; }
  else if (frag < FS_MF) { W = mw2; K = 128; Nv = 128; start = FS_M2; }
  else if (frag < FS_P0) { W = mw3; K = 128; Nv = 75;  start = FS_MF; }
  else if (frag < FS_P1) { W = pw0; K = 96;  Nv = 128; start = FS_P0; }
  else if (frag < FS_P2) { W = pw1; K = 128; Nv = 128; start = FS_P1; }
  else                   { W = pw2; K = 128; Nv = 128; start = FS_P2; }

  const int lf    = frag - start;
  const int NK    = K >> 5;
  const int ntile = lf / NK;
  const int kc    = lf % NK;
  const int col   = ntile * 16 + (lane & 15);
  const int kh8   = (lane >> 4) * 8;
  const bool valid = (col < Nv);

  v16h v;
  #pragma unroll
  for (int j = 0; j < 8; ++j) {
    const int k = kc * 32 + kh8 + 2 * j + (j >= 4 ? 8 : 0);
    v[2*j]   = valid ? (_Float16)W[(size_t)(k)   * Nv + col] : (_Float16)0.0f;
    v[2*j+1] = valid ? (_Float16)W[(size_t)(k+1) * Nv + col] : (_Float16)0.0f;
  }
  *(v16h*)(wp + ((size_t)frag * 512 + lane * 16)) = v;
}

// ---------------------------------------------------------------------------
// Prep 2: convert concat(s,g) to f16 rows of 96 halves (contiguous, 192B/row)
// so the main kernel can DMA tiles with the Tensor Data Mover.
// ---------------------------------------------------------------------------
__global__ void __launch_bounds__(256)
pack_inputs_kernel(const float* __restrict__ s, const float* __restrict__ g,
                   _Float16* __restrict__ in16, int B)
{
  const int idx = blockIdx.x * 256 + threadIdx.x;  // one per 8 halves
  if (idx >= B * 12) return;
  const int r  = idx / 12;
  const int c8 = (idx % 12) * 8;
  v4f f0, f1;
  if (c8 < 64) {
    const float* p = s + (size_t)r * 64 + c8;
    f0 = *(const v4f*)p; f1 = *(const v4f*)(p + 4);
  } else {
    const float* p = g + (size_t)r * 32 + (c8 - 64);
    f0 = *(const v4f*)p; f1 = *(const v4f*)(p + 4);
  }
  v8h h;
  #pragma unroll
  for (int j = 0; j < 4; ++j) { h[j] = (_Float16)f0[j]; h[4+j] = (_Float16)f1[j]; }
  *(v8h*)(in16 + (size_t)r * 96 + c8) = h;
}

// Build A fragment (16x32 f16, ISA layout) from LDS: two 16B aligned reads.
__device__ __forceinline__ v16h make_a_frag(const _Float16* p) {
  v8h lo = *(const v8h*)p;
  v8h hi = *(const v8h*)(p + 16);
  return __builtin_shufflevector(lo, hi, 0,1,2,3,4,5,6,7,8,9,10,11,12,13,14,15);
}

// One 128-wide hidden layer via WMMA; B fragments pre-packed in wp.
// src K-chunks live at (chunk0+kc)*cstride; dst is H layout (contiguous 128).
template<int K>
__device__ __forceinline__ void layer_h(const _Float16* __restrict__ src, int sstride,
                                        int chunk0, int cstride,
                                        _Float16* __restrict__ dst,
                                        const _Float16* __restrict__ wp, int frag_start,
                                        const float* __restrict__ bias,
                                        bool relu, int tid)
{
  constexpr int NK = K / 32;
  const int lane = tid & 31;
  const int wv   = tid >> 5;
  const int col  = lane & 15;          // also A-row within m-tile
  const int kh8  = (lane >> 4) * 8;
  const int cgl  = wv * 16 + col;

  const _Float16* wbase = wp + ((size_t)(frag_start + wv * NK) * 512) + lane * 16;
  v16h bf[NK];
  #pragma unroll
  for (int kc = 0; kc < NK; ++kc)
    bf[kc] = *(const v16h*)(wbase + (size_t)kc * 512);

  const float bv = bias[cgl];
  const _Float16* pa0 = src + col * sstride + chunk0 * cstride + kh8;
  _Float16*       pd0 = dst + kh8 * H_STRIDE + cgl;

  #pragma unroll
  for (int mt = 0; mt < 4; ++mt) {
    v8f acc;
    #pragma unroll
    for (int r = 0; r < 8; ++r) acc[r] = bv;
    #pragma unroll
    for (int kc = 0; kc < NK; ++kc) {
      v16h a = make_a_frag(pa0 + mt * 16 * sstride + kc * cstride);
      acc = __builtin_amdgcn_wmma_f32_16x16x32_f16(false, a, false, bf[kc],
                                                   (short)0, acc, false, false);
    }
    #pragma unroll
    for (int r = 0; r < 8; ++r) {
      float v = acc[r];
      if (relu) v = fmaxf(v, 0.0f);
      pd0[(mt * 16 + r) * H_STRIDE] = (_Float16)v;
    }
  }
}

// MDN head: mu[64][75] = src[64][128] @ W[128][75] + b (f16 out, stride 80).
__device__ __forceinline__ void mdn_final(const _Float16* __restrict__ src,
                                          _Float16* __restrict__ mu,
                                          const _Float16* __restrict__ wp,
                                          const float* __restrict__ bias, int tid)
{
  const int lane = tid & 31;
  const int wv   = tid >> 5;
  if (wv >= 5) return;                 // waves 5..7 idle (wave-uniform branch)
  const int col  = lane & 15;
  const int kh8  = (lane >> 4) * 8;
  const int cgl  = wv * 16 + col;
  const bool valid = (cgl < 75);

  const _Float16* wbase = wp + ((size_t)(FS_MF + wv * 4) * 512) + lane * 16;
  v16h bf[4];
  #pragma unroll
  for (int kc = 0; kc < 4; ++kc)
    bf[kc] = *(const v16h*)(wbase + (size_t)kc * 512);

  const float bv = valid ? bias[cgl] : 0.0f;
  const _Float16* pa0 = src + col * H_STRIDE + kh8;
  _Float16*       pd0 = mu + kh8 * MU_STRIDE + cgl;

  #pragma unroll
  for (int mt = 0; mt < 4; ++mt) {
    v8f acc;
    #pragma unroll
    for (int r = 0; r < 8; ++r) acc[r] = bv;
    #pragma unroll
    for (int kc = 0; kc < 4; ++kc) {
      v16h a = make_a_frag(pa0 + mt * 16 * H_STRIDE + kc * H_CSTRIDE);
      acc = __builtin_amdgcn_wmma_f32_16x16x32_f16(false, a, false, bf[kc],
                                                   (short)0, acc, false, false);
    }
    #pragma unroll
    for (int r = 0; r < 8; ++r)
      if (valid) pd0[(mt * 16 + r) * MU_STRIDE] = (_Float16)acc[r];
  }
}

// Tiny 128->3 head; weights pre-staged transposed in LDS (w3t[c][k]).
__device__ __forceinline__ void final3(const _Float16* __restrict__ src,
                                       const float* __restrict__ w3t,
                                       const float* __restrict__ b,
                                       float* __restrict__ dst, int tid)
{
  if (tid < 192) {
    const int r = tid / 3, c = tid % 3;
    float acc = b[c];
    #pragma unroll
    for (int k8 = 0; k8 < 128; k8 += 8) {
      v8h h  = *(const v8h*)(src + r * H_STRIDE + k8);
      v4f w0 = *(const v4f*)(w3t + c * W3_STRIDE + k8);
      v4f w1 = *(const v4f*)(w3t + c * W3_STRIDE + k8 + 4);
      acc = fmaf((float)h[0], w0[0], acc);
      acc = fmaf((float)h[1], w0[1], acc);
      acc = fmaf((float)h[2], w0[2], acc);
      acc = fmaf((float)h[3], w0[3], acc);
      acc = fmaf((float)h[4], w1[0], acc);
      acc = fmaf((float)h[5], w1[1], acc);
      acc = fmaf((float)h[6], w1[2], acc);
      acc = fmaf((float)h[7], w1[3], acc);
    }
    dst[r*3 + c] = acc;
  }
}

__global__ void __launch_bounds__(256)
fused_mlp_kde_kernel(
    const float* eb0, const float* eb1, const float* eb2,
    const float* ew3, const float* eb3,
    const float* pb0, const float* pb1, const float* pb2,
    const float* pw3, const float* pb3,
    const float* mb0, const float* mb1, const float* mb2, const float* mb3,
    const _Float16* __restrict__ wp, const _Float16* __restrict__ in16,
    float* __restrict__ out_base, float* __restrict__ gv, float* __restrict__ partials)
{
  __shared__ __align__(16) _Float16 s_in[TILE_ROWS * IN_STRIDE];
  __shared__ __align__(16) _Float16 s_a [TILE_ROWS * H_STRIDE];
  __shared__ __align__(16) _Float16 s_b [TILE_ROWS * H_STRIDE];
  __shared__ __align__(16) _Float16 s_mu[TILE_ROWS * MU_STRIDE];
  __shared__ __align__(16) float s_w3 [3 * W3_STRIDE];
  __shared__ float s_z  [TILE_ROWS * 3];
  __shared__ float s_ail[TILE_ROWS * 3];
  __shared__ float s_red[TILE_ROWS];

  const int tid  = threadIdx.x;
  const int row0 = blockIdx.x * TILE_ROWS;

  // ---- Tensor Data Mover: DMA 64x96 f16 tile (48 dwords/row) global -> LDS.
  // D# per CDNA5 ISA 8.3-8.5. Padding: +4 dwords after every 16 dwords
  // => LDS row stride 60 dwords (conflict-free), segments 32h @ 40h apart.
  if (tid < 32) {   // wave 0 issues the TDM op (EXEC ignored by TDM)
    const unsigned lds_off = (unsigned)(unsigned long long)(uintptr_t)s_in;
    const unsigned long long ga =
        (unsigned long long)(uintptr_t)(in16 + (size_t)row0 * 96);
    u32x4 g0;
    g0[0] = 1u;                                   // count=1 (valid descriptor)
    g0[1] = lds_off;                              // lds_addr (bytes)
    g0[2] = (unsigned)(ga & 0xFFFFFFFFu);         // global_addr[31:0]
    g0[3] = (unsigned)((ga >> 32) & 0x01FFFFFFu)  // global_addr[56:32]
          | (2u << 30);                           // type=2 ("image")
    i32x8 g1;
    g1[0] = (2 << 16)      // data_size = 4B elements
          | (1 << 20)      // pad_enable
          | (3 << 22)      // pad_interval: 16 dwords
          | (3 << 25);     // pad_amount:   4 dwords
    g1[1] = 48 << 16;      // tensor_dim0 = 48 (bits 79:64 are in next word)
    g1[2] = 64 << 16;      // tensor_dim1[15:0] = 64 rows (bits 95:80)
    g1[3] = 48 << 16;      // tile_dim0 = 48 (bits 127:112)
    g1[4] = 64;            // tile_dim1 = 64 (bits 143:128); tile_dim2 = 0
    g1[5] = 48;            // tensor_dim0_stride = 48
    g1[6] = 0; g1[7] = 0;  // tensor_dim1_stride = 0 (unused, 2D)
    i32x4 gz = {0, 0, 0, 0};
    i32x8 gz8 = {0, 0, 0, 0, 0, 0, 0, 0};
    __builtin_amdgcn_tensor_load_to_lds(g0, g1, gz, gz, gz8, 0);
    __builtin_amdgcn_s_wait_tensorcnt(0);
  }
  __syncthreads();

  // ---- encoder: s -> z
  layer_h<64 >(s_in, IN_STRIDE, 0, IN_CSTRIDE, s_a, wp, FS_E0, eb0, true, tid); __syncthreads();
  layer_h<128>(s_a,  H_STRIDE,  0, H_CSTRIDE,  s_b, wp, FS_E1, eb1, true, tid); __syncthreads();
  layer_h<128>(s_b,  H_STRIDE,  0, H_CSTRIDE,  s_a, wp, FS_E2, eb2, true, tid); __syncthreads();
  for (int i = tid; i < 384; i += 256)
    s_w3[(i >> 7) * W3_STRIDE + (i & 127)] = ew3[(i & 127) * 3 + (i >> 7)];
  __syncthreads();
  final3(s_a, s_w3, eb3, s_z, tid);                                             __syncthreads();

  // ---- MDN: g -> mu[25*3]
  layer_h<32 >(s_in, IN_STRIDE, 2, IN_CSTRIDE, s_a, wp, FS_M0, mb0, true, tid); __syncthreads();
  layer_h<128>(s_a,  H_STRIDE,  0, H_CSTRIDE,  s_b, wp, FS_M1, mb1, true, tid); __syncthreads();
  layer_h<128>(s_b,  H_STRIDE,  0, H_CSTRIDE,  s_a, wp, FS_M2, mb2, true, tid); __syncthreads();
  mdn_final(s_a, s_mu, wp, mb3, tid);                                           __syncthreads();

  // ---- policy: concat(s,g) -> a_il
  layer_h<96 >(s_in, IN_STRIDE, 0, IN_CSTRIDE, s_a, wp, FS_P0, pb0, true, tid); __syncthreads();
  layer_h<128>(s_a,  H_STRIDE,  0, H_CSTRIDE,  s_b, wp, FS_P1, pb1, true, tid); __syncthreads();
  layer_h<128>(s_b,  H_STRIDE,  0, H_CSTRIDE,  s_a, wp, FS_P2, pb2, true, tid); __syncthreads();
  for (int i = tid; i < 384; i += 256)
    s_w3[(i >> 7) * W3_STRIDE + (i & 127)] = pw3[(i & 127) * 3 + (i >> 7)];
  __syncthreads();
  final3(s_a, s_w3, pb3, s_ail, tid);                                           __syncthreads();

  // ---- KDE epilogue (one thread per row)
  if (tid < TILE_ROWS) {
    const int r = tid;
    const float z0 = s_z[r*3+0], z1 = s_z[r*3+1], z2 = s_z[r*3+2];
    float rho = 0.f, g0 = 0.f, g1 = 0.f, g2 = 0.f;
    const float cnorm = 0.3989422804014327f;   // (2*pi*h^3)^-0.5, h=1
    #pragma unroll
    for (int i = 0; i < 25; ++i) {
      const float d0 = z0 - (float)s_mu[r*MU_STRIDE + 3*i + 0];
      const float d1 = z1 - (float)s_mu[r*MU_STRIDE + 3*i + 1];
      const float d2 = z2 - (float)s_mu[r*MU_STRIDE + 3*i + 2];
      const float p  = cnorm * __expf(-0.5f * (d0*d0 + d1*d1 + d2*d2));
      rho += p; g0 -= p*d0; g1 -= p*d1; g2 -= p*d2;
    }
    if (g0 != g0) g0 = 0.f;  // nan_to_num
    if (g1 != g1) g1 = 0.f;
    if (g2 != g2) g2 = 0.f;
    const float pm = tanhf(rho * 0.002f);
    const float om = 1.0f - pm;
    const size_t o = (size_t)(row0 + r) * 3;
    out_base[o+0] = pm * s_ail[r*3+0];
    out_base[o+1] = pm * s_ail[r*3+1];
    out_base[o+2] = pm * s_ail[r*3+2];
    gv[o+0] = om * g0;  gv[o+1] = om * g1;  gv[o+2] = om * g2;
    s_red[r] = g0*g0 + g1*g1 + g2*g2;      // raw-grad sumsq for global norm
  }
  __syncthreads();
  if (tid == 0) {
    float t = 0.f;
    #pragma unroll 8
    for (int i = 0; i < TILE_ROWS; ++i) t += s_red[i];
    partials[blockIdx.x] = t;
  }
}

__global__ void __launch_bounds__(256)
reduce_norm_kernel(const float* __restrict__ partials, int n, float* __restrict__ scale)
{
  __shared__ float red[256];
  float t = 0.f;
  for (int i = threadIdx.x; i < n; i += 256) t += partials[i];
  red[threadIdx.x] = t;
  __syncthreads();
  for (int sft = 128; sft > 0; sft >>= 1) {
    if (threadIdx.x < sft) red[threadIdx.x] += red[threadIdx.x + sft];
    __syncthreads();
  }
  if (threadIdx.x == 0) {
    const float nrm = sqrtf(red[0]);
    scale[0] = (nrm > 0.f) ? (0.0005f / nrm) : 0.f;   // NI / ||grad||_F
  }
}

__global__ void __launch_bounds__(256)
finalize_kernel(float* __restrict__ out, const float* __restrict__ gv,
                const float* __restrict__ scale, int n)
{
  const int i = blockIdx.x * 256 + threadIdx.x;
  if (i < n) out[i] = out[i] + gv[i] * scale[0];
}

extern "C" void kernel_launch(void* const* d_in, const int* in_sizes, int n_in,
                              void* d_out, int out_size, void* d_ws, size_t ws_size,
                              hipStream_t stream)
{
  (void)n_in; (void)out_size; (void)ws_size;
  const float* s = (const float*)d_in[0];
  const float* g = (const float*)d_in[1];
  const float* W[24];
  for (int i = 0; i < 24; ++i) W[i] = (const float*)d_in[2 + i];
  // W: [0..7]=e w0,b0..w3,b3  [8..15]=p  [16..23]=m

  const int B    = in_sizes[0] / 64;     // s is [B,64]
  const int nblk = B / TILE_ROWS;        // 64 rows per block

  // workspace: packed weights | f16 inputs | gv | partials | scale
  _Float16* wp     = (_Float16*)d_ws;
  _Float16* in16   = (_Float16*)((char*)d_ws + WP_BYTES);
  float* gv        = (float*)((char*)d_ws + WP_BYTES + (size_t)B * 96 * 2);
  float* partials  = gv + (size_t)B * 3;
  float* scale     = partials + nblk;
  float* out       = (float*)d_out;

  pack_weights_kernel<<<(NFRAG * 32 + 255) / 256, 256, 0, stream>>>(
      W[0], W[2], W[4],            // ew0, ew1, ew2
      W[16], W[18], W[20], W[22],  // mw0, mw1, mw2, mw3
      W[8], W[10], W[12],          // pw0, pw1, pw2
      wp);

  pack_inputs_kernel<<<(B * 12 + 255) / 256, 256, 0, stream>>>(s, g, in16, B);

  fused_mlp_kde_kernel<<<nblk, 256, 0, stream>>>(
      W[1], W[3], W[5],            // eb0, eb1, eb2
      W[6], W[7],                  // ew3, eb3
      W[9], W[11], W[13],          // pb0, pb1, pb2
      W[14], W[15],                // pw3, pb3
      W[17], W[19], W[21], W[23],  // mb0, mb1, mb2, mb3
      wp, in16, out, gv, partials);

  reduce_norm_kernel<<<1, 256, 0, stream>>>(partials, nblk, scale);

  const int n = B * 3;
  finalize_kernel<<<(n + 255) / 256, 256, 0, stream>>>(out, gv, scale, n);
}
